// RelPositionMultiHeadedAttentionChunk_6717328851577
// MI455X (gfx1250) — compile-verified, matrix operand order checked by reading, not run on hardware
//
#include <hip/hip_runtime.h>
#include <hip/hip_bf16.h>

typedef _Float16 half_t;
typedef __attribute__((ext_vector_type(16))) _Float16 v16h;
typedef __attribute__((ext_vector_type(8)))  _Float16 v8h;
typedef __attribute__((ext_vector_type(4)))  _Float16 v4h;
typedef __attribute__((ext_vector_type(8)))  float    v8f;

#define NUM_HEADS 8
#define EMBED 512
#define D_K 64
#define BATCH 16
#define T1 512
#define T2 512
#define N_POS 1023
#define NPAD 1024

// ---------------------------------------------------------------------------
// Fragment loaders. CDNA5 16-bit A-matrix 16x32 layout: lane (lr = lane&15,
// hi = lane>>4) holds row M=lr, K = {hi*8 .. hi*8+7} and {hi*8+16 .. hi*8+23}.
// B mirrors with N=lr. Both are two contiguous 8-half (16B) chunks per lane.
// ---------------------------------------------------------------------------
__device__ __forceinline__ v16h load_h16(const half_t* p) {
    v8h lo = *(const v8h*)(p);
    v8h hi = *(const v8h*)(p + 16);
    v16h r;
#pragma unroll
    for (int t = 0; t < 8; ++t) { r[t] = lo[t]; r[8 + t] = hi[t]; }
    return r;
}

__device__ __forceinline__ v8f wmma16(v16h a, v16h b, v8f c) {
    return __builtin_amdgcn_wmma_f32_16x16x32_f16(false, a, false, b,
                                                  (short)0, c, false, false);
}

__device__ __forceinline__ v8f vzero8() {
    v8f z = {0.f, 0.f, 0.f, 0.f, 0.f, 0.f, 0.f, 0.f};
    return z;
}

// ---------------------------------------------------------------------------
// Elementwise f32 -> f16 conversion (one pass; removes cvt from GEMM loops).
// 4 elements per thread.
// ---------------------------------------------------------------------------
__global__ __launch_bounds__(256) void k_cvt(const float* __restrict__ src,
                                             half_t* __restrict__ dst,
                                             size_t n4) {
    size_t t = (size_t)blockIdx.x * 256 + threadIdx.x;
    if (t >= n4) return;
    const float4 f = *(const float4*)(src + t * 4);
    v4h h;
    h[0] = (_Float16)f.x; h[1] = (_Float16)f.y;
    h[2] = (_Float16)f.z; h[3] = (_Float16)f.w;
    *(v4h*)(dst + t * 4) = h;
}

// ---------------------------------------------------------------------------
// K1: Y = X16(f16, Mx512) @ W16(f16, 512x512)^T + bias.
// Block = 8 waves sharing one 16-row A strip (staged once in LDS, read back
// with ds_load_b128); wave w covers N slice [w*64, w*64+64).
// Optional epilogues: f16 outA = Y + pbA[col], f16 outB = Y + pbB[col],
// f32 outF = Y.
// ---------------------------------------------------------------------------
__global__ __launch_bounds__(256) void k_proj_f16(const half_t* __restrict__ X16,
                                                  const half_t* __restrict__ W16,
                                                  const float* __restrict__ bias,
                                                  half_t* __restrict__ outA,
                                                  const float* __restrict__ pbA,
                                                  half_t* __restrict__ outB,
                                                  const float* __restrict__ pbB,
                                                  float* __restrict__ outF,
                                                  int M) {
    __shared__ half_t Asmem[16 * EMBED];     // 16 KiB f16 A strip

    const int lane = threadIdx.x & 31, warp = threadIdx.x >> 5;
    const int mtile = blockIdx.x;            // one block per 16-row strip
    const int lr = lane & 15, hi = lane >> 4;
    const int n0 = warp * 64;

    // Stage A strip: 16 rows x 512 halves = 16384 B, contiguous in memory.
    {
        const uint4* g = (const uint4*)(X16 + (size_t)mtile * 16 * EMBED) +
                         (size_t)threadIdx.x * 4;
        uint4* s = (uint4*)(Asmem) + (size_t)threadIdx.x * 4;
        s[0] = g[0]; s[1] = g[1]; s[2] = g[2]; s[3] = g[3];
    }
    __syncthreads();

    v8f acc[4];
#pragma unroll
    for (int nt = 0; nt < 4; ++nt) acc[nt] = vzero8();

    for (int k0 = 0; k0 < EMBED; k0 += 32) {
        v16h a = load_h16(Asmem + (size_t)lr * EMBED + k0 + hi * 8);
        if (k0 + 32 < EMBED) {
            // Prefetch next K-step of B (lowers to global_prefetch_b8).
            __builtin_prefetch(W16 + (size_t)(n0 + lr) * EMBED + k0 + 32, 0, 1);
        }
#pragma unroll
        for (int nt = 0; nt < 4; ++nt) {
            const int col = n0 + nt * 16 + lr;
            v16h bf = load_h16(W16 + (size_t)col * EMBED + k0 + hi * 8);
            acc[nt] = wmma16(a, bf, acc[nt]);
        }
    }

#pragma unroll
    for (int nt = 0; nt < 4; ++nt) {
        const int col = n0 + nt * 16 + lr;
        const float bv = bias ? bias[col] : 0.f;
        const float ua = pbA ? pbA[col] : 0.f;
        const float ub = pbB ? pbB[col] : 0.f;
#pragma unroll
        for (int r = 0; r < 8; ++r) {
            const int m = mtile * 16 + r + hi * 8;
            const float y = acc[nt][r] + bv;
            if (outF) outF[(size_t)m * EMBED + col] = y;
            if (outA) outA[(size_t)m * EMBED + col] = (half_t)(y + ua);
            if (outB) outB[(size_t)m * EMBED + col] = (half_t)(y + ub);
        }
    }
    (void)M;
}

// ---------------------------------------------------------------------------
// Transpose V (b,t,h*64+d) -> VT (b,h,d,t) so attn@V B-fragments are
// contiguous along K=t.
// ---------------------------------------------------------------------------
__global__ __launch_bounds__(256) void k_vt(const half_t* __restrict__ v,
                                            half_t* __restrict__ vt) {
    size_t idx = (size_t)blockIdx.x * 256 + threadIdx.x;
    if (idx >= (size_t)BATCH * NUM_HEADS * D_K * T2) return;
    const int j = idx & (T2 - 1);
    size_t r = idx >> 9;
    const int d = r & (D_K - 1);
    r >>= 6;
    const int h = r & (NUM_HEADS - 1);
    const int b = (int)(r >> 3);
    vt[idx] = v[(((size_t)b * T2 + j) * EMBED) + h * D_K + d];
}

// ---------------------------------------------------------------------------
// K2: matrix_ac[b,h,i,j] = sum_d (q+u)[b,i,h,d] * k[b,j,h,d]   (K=64)
// ---------------------------------------------------------------------------
__global__ __launch_bounds__(256) void k_scores_ac(const half_t* __restrict__ qu,
                                                   const half_t* __restrict__ kk,
                                                   float* __restrict__ sc) {
    const int lane = threadIdx.x & 31, warp = threadIdx.x >> 5;
    const int wid = blockIdx.x * 8 + warp;            // 32768 waves total
    const int jt = wid & 7;
    const int it = (wid >> 3) & 31;
    const int h  = (wid >> 8) & 7;
    const int b  = wid >> 11;
    const int lr = lane & 15, hi = lane >> 4;

    const half_t* abase = qu + ((size_t)(b * T1 + it * 16 + lr)) * EMBED + h * D_K;
    v8f acc[4];
#pragma unroll
    for (int nt = 0; nt < 4; ++nt) acc[nt] = vzero8();

#pragma unroll
    for (int k0 = 0; k0 < D_K; k0 += 32) {
        v16h a = load_h16(abase + k0 + hi * 8);
#pragma unroll
        for (int nt = 0; nt < 4; ++nt) {
            const int j = jt * 64 + nt * 16 + lr;
            v16h bf = load_h16(kk + ((size_t)(b * T2 + j)) * EMBED + h * D_K + k0 + hi * 8);
            acc[nt] = wmma16(a, bf, acc[nt]);
        }
    }

    float* srow = sc + ((size_t)(b * NUM_HEADS + h)) * T1 * T2;
#pragma unroll
    for (int nt = 0; nt < 4; ++nt) {
        const int j = jt * 64 + nt * 16 + lr;
#pragma unroll
        for (int r = 0; r < 8; ++r) {
            const int i = it * 16 + r + hi * 8;
            srow[(size_t)i * T2 + j] = acc[nt][r];
        }
    }
}

// ---------------------------------------------------------------------------
// K3 (per batch): matrix_bd[h,i,n] = sum_d (q+v)[i,h,d] * p[n,h,d], n padded
// to 1024; column n=1023 is never read by the shift (range is [0,1022]).
// ---------------------------------------------------------------------------
__global__ __launch_bounds__(256) void k_scores_bd(const half_t* __restrict__ qv_b,
                                                   const half_t* __restrict__ p_b,
                                                   float* __restrict__ bd) {
    const int lane = threadIdx.x & 31, warp = threadIdx.x >> 5;
    const int wid = blockIdx.x * 8 + warp;            // 4096 waves
    const int jt = wid & 15;                          // 16 tiles of 64 -> 1024
    const int it = (wid >> 4) & 31;
    const int h  = wid >> 9;
    const int lr = lane & 15, hi = lane >> 4;

    const half_t* abase = qv_b + (size_t)(it * 16 + lr) * EMBED + h * D_K;
    v8f acc[4];
#pragma unroll
    for (int nt = 0; nt < 4; ++nt) acc[nt] = vzero8();

#pragma unroll
    for (int k0 = 0; k0 < D_K; k0 += 32) {
        v16h a = load_h16(abase + k0 + hi * 8);
#pragma unroll
        for (int nt = 0; nt < 4; ++nt) {
            int n = jt * 64 + nt * 16 + lr;
            int prow = n < (N_POS - 1) ? n : (N_POS - 1);   // clamp OOB pads
            v16h bf = load_h16(p_b + (size_t)prow * EMBED + h * D_K + k0 + hi * 8);
            acc[nt] = wmma16(a, bf, acc[nt]);
        }
    }

#pragma unroll
    for (int nt = 0; nt < 4; ++nt) {
        const int n = jt * 64 + nt * 16 + lr;
#pragma unroll
        for (int r = 0; r < 8; ++r) {
            const int i = it * 16 + r + hi * 8;
            bd[((size_t)h * T1 + i) * NPAD + n] = acc[nt][r];
        }
    }
}

// ---------------------------------------------------------------------------
// K4 (per batch): rel-shift + mask + scale + softmax; writes f16 attn.
// One 256-thread block per (h,i) row; 2 elements/thread.
// ---------------------------------------------------------------------------
__global__ __launch_bounds__(256) void k_softmax(const float* __restrict__ sc_b,
                                                 const float* __restrict__ bd,
                                                 const unsigned char* __restrict__ mask_b,
                                                 const unsigned char* __restrict__ chunk,
                                                 half_t* __restrict__ att_b) {
    const int h = blockIdx.x >> 9;
    const int i = blockIdx.x & 511;
    const int lane = threadIdx.x & 31, warp = threadIdx.x >> 5;
    const float* srow  = sc_b + ((size_t)h * T1 + i) * T2;
    // bd index for column j is (j - i + 511): offset the row pointer once.
    const float* bdrow = bd + ((size_t)h * T1 + i) * NPAD + (T1 - 1 - i);

    const int j0 = threadIdx.x;
    const int j1 = threadIdx.x + 256;
    const bool m0 = chunk[(size_t)i * T2 + j0] || mask_b[j0];
    const bool m1 = chunk[(size_t)i * T2 + j1] || mask_b[j1];
    const float NEG = -3.4028235e38f;
    float v0 = m0 ? NEG : (srow[j0] + bdrow[j0]) * 0.125f;   // 1/sqrt(64)
    float v1 = m1 ? NEG : (srow[j1] + bdrow[j1]) * 0.125f;

    __shared__ float smax[8];
    __shared__ float ssum[8];

    float mx = fmaxf(v0, v1);
#pragma unroll
    for (int off = 16; off > 0; off >>= 1) mx = fmaxf(mx, __shfl_xor(mx, off, 32));
    if (lane == 0) smax[warp] = mx;
    __syncthreads();
    float rowmax = smax[0];
#pragma unroll
    for (int w = 1; w < 8; ++w) rowmax = fmaxf(rowmax, smax[w]);

    float e0 = m0 ? 0.f : __expf(v0 - rowmax);
    float e1 = m1 ? 0.f : __expf(v1 - rowmax);
    float s = e0 + e1;
#pragma unroll
    for (int off = 16; off > 0; off >>= 1) s += __shfl_xor(s, off, 32);
    if (lane == 0) ssum[warp] = s;
    __syncthreads();
    float total = 0.f;
#pragma unroll
    for (int w = 0; w < 8; ++w) total += ssum[w];
    const float inv = 1.f / fmaxf(total, 1e-30f);

    half_t* arow = att_b + ((size_t)h * T1 + i) * T2;
    arow[j0] = (half_t)(e0 * inv);
    arow[j1] = (half_t)(e1 * inv);
}

// ---------------------------------------------------------------------------
// K5: ctx[b,i,h*64+d] = sum_j attn[b,h,i,j] * v[b,j,h,d]  (K=512, f16 out so
// the final projection runs the pure-f16 GEMM path).
// ---------------------------------------------------------------------------
__global__ __launch_bounds__(256) void k_ctx(const half_t* __restrict__ att,
                                             const half_t* __restrict__ vt,
                                             half_t* __restrict__ ctx) {
    const int lane = threadIdx.x & 31, warp = threadIdx.x >> 5;
    const int wid = blockIdx.x * 8 + warp;            // 4096 waves
    const int it = wid & 31;
    const int h  = (wid >> 5) & 7;
    const int b  = wid >> 8;
    const int lr = lane & 15, hi = lane >> 4;

    const half_t* abase = att + (((size_t)(b * NUM_HEADS + h) * T1) + it * 16 + lr) * T2;
    const half_t* bbase = vt + ((size_t)(b * NUM_HEADS + h) * D_K) * T2;
    v8f acc[4];
#pragma unroll
    for (int nt = 0; nt < 4; ++nt) acc[nt] = vzero8();

    for (int j0 = 0; j0 < T2; j0 += 32) {
        v16h a = load_h16(abase + j0 + hi * 8);
#pragma unroll
        for (int nt = 0; nt < 4; ++nt) {
            const int d = nt * 16 + lr;
            v16h bf = load_h16(bbase + (size_t)d * T2 + j0 + hi * 8);
            acc[nt] = wmma16(a, bf, acc[nt]);
        }
    }

#pragma unroll
    for (int nt = 0; nt < 4; ++nt) {
        const int d = nt * 16 + lr;
#pragma unroll
        for (int r = 0; r < 8; ++r) {
            const int i = it * 16 + r + hi * 8;
            ctx[((size_t)b * T1 + i) * EMBED + h * D_K + d] = (half_t)acc[nt][r];
        }
    }
}

// ---------------------------------------------------------------------------
// Host launcher
// ---------------------------------------------------------------------------
extern "C" void kernel_launch(void* const* d_in, const int* in_sizes, int n_in,
                              void* d_out, int out_size, void* d_ws, size_t ws_size,
                              hipStream_t stream) {
    (void)in_sizes; (void)n_in; (void)out_size; (void)ws_size;

    const float* query   = (const float*)d_in[0];
    const float* key     = (const float*)d_in[1];
    const float* value   = (const float*)d_in[2];
    const float* pos_enc = (const float*)d_in[3];
    const unsigned char* mask  = (const unsigned char*)d_in[4];
    const unsigned char* chunk = (const unsigned char*)d_in[5];
    const float* Wq = (const float*)d_in[6];
    const float* bq = (const float*)d_in[7];
    const float* Wk = (const float*)d_in[8];
    const float* bk = (const float*)d_in[9];
    const float* Wv = (const float*)d_in[10];
    const float* bv = (const float*)d_in[11];
    const float* Wpos = (const float*)d_in[12];
    const float* Wo = (const float*)d_in[13];
    const float* bo = (const float*)d_in[14];
    const float* pbu = (const float*)d_in[15];
    const float* pbv = (const float*)d_in[16];
    float* out = (float*)d_out;

    // Workspace carve-up (bytes, all 256-aligned)
    char* ws = (char*)d_ws;
    const size_t SZ_ACT16 = (size_t)BATCH * T1 * EMBED * sizeof(half_t);        // 8 MiB
    const size_t SZ_P16   = (size_t)BATCH * N_POS * EMBED * sizeof(half_t);     // 16 MiB
    const size_t SZ_W16   = (size_t)EMBED * EMBED * sizeof(half_t);             // 512 KiB
    const size_t SZ_SC    = (size_t)BATCH * NUM_HEADS * T1 * T2 * sizeof(float);// 128 MiB
    const size_t SZ_BD    = (size_t)NUM_HEADS * T1 * NPAD * sizeof(float);      // 16 MiB
    const size_t SZ_ATT   = (size_t)BATCH * NUM_HEADS * T1 * T2 * sizeof(half_t);
    size_t off = 0;
    half_t* Qin16 = (half_t*)(ws + off); off += SZ_ACT16;   // f16 inputs
    half_t* Kin16 = (half_t*)(ws + off); off += SZ_ACT16;
    half_t* Vin16 = (half_t*)(ws + off); off += SZ_ACT16;
    half_t* Pin16 = (half_t*)(ws + off); off += SZ_P16;
    half_t* Wq16  = (half_t*)(ws + off); off += SZ_W16;     // f16 weights
    half_t* Wk16  = (half_t*)(ws + off); off += SZ_W16;
    half_t* Wv16  = (half_t*)(ws + off); off += SZ_W16;
    half_t* Wp16  = (half_t*)(ws + off); off += SZ_W16;
    half_t* Wo16  = (half_t*)(ws + off); off += SZ_W16;
    half_t* QU16  = (half_t*)(ws + off); off += SZ_ACT16;   // projected f16
    half_t* QV16  = (half_t*)(ws + off); off += SZ_ACT16;
    half_t* K16   = (half_t*)(ws + off); off += SZ_ACT16;
    half_t* V16   = (half_t*)(ws + off); off += SZ_ACT16;
    half_t* P16   = (half_t*)(ws + off); off += SZ_P16;
    half_t* VT16  = (half_t*)(ws + off); off += SZ_ACT16;
    float*  SC    = (float*)(ws + off);  off += SZ_SC;
    float*  BD    = (float*)(ws + off);  off += SZ_BD;
    half_t* ATT   = (half_t*)(ws + off); off += SZ_ATT;
    half_t* CTX16 = (half_t*)(ws + off); off += SZ_ACT16;

    const int M_qkv = BATCH * T1;          // 8192
    const int M_pos = BATCH * N_POS;       // 16368 (multiple of 16)

    // One-time f32 -> f16 conversions (memory-bound, single pass).
    {
        const size_t n_act = (size_t)BATCH * T1 * EMBED / 4;
        const size_t n_pos = (size_t)BATCH * N_POS * EMBED / 4;
        const size_t n_w   = (size_t)EMBED * EMBED / 4;
        k_cvt<<<(unsigned)((n_act + 255) / 256), 256, 0, stream>>>(query, Qin16, n_act);
        k_cvt<<<(unsigned)((n_act + 255) / 256), 256, 0, stream>>>(key,   Kin16, n_act);
        k_cvt<<<(unsigned)((n_act + 255) / 256), 256, 0, stream>>>(value, Vin16, n_act);
        k_cvt<<<(unsigned)((n_pos + 255) / 256), 256, 0, stream>>>(pos_enc, Pin16, n_pos);
        k_cvt<<<(unsigned)((n_w + 255) / 256), 256, 0, stream>>>(Wq, Wq16, n_w);
        k_cvt<<<(unsigned)((n_w + 255) / 256), 256, 0, stream>>>(Wk, Wk16, n_w);
        k_cvt<<<(unsigned)((n_w + 255) / 256), 256, 0, stream>>>(Wv, Wv16, n_w);
        k_cvt<<<(unsigned)((n_w + 255) / 256), 256, 0, stream>>>(Wpos, Wp16, n_w);
        k_cvt<<<(unsigned)((n_w + 255) / 256), 256, 0, stream>>>(Wo, Wo16, n_w);
    }

    // Projections (pure-f16 WMMA, LDS-staged A). Q fuses both pos-bias adds.
    k_proj_f16<<<M_qkv / 16, 256, 0, stream>>>(Qin16, Wq16, bq, QU16, pbu, QV16, pbv, nullptr, M_qkv);
    k_proj_f16<<<M_qkv / 16, 256, 0, stream>>>(Kin16, Wk16, bk, K16, nullptr, nullptr, nullptr, nullptr, M_qkv);
    k_proj_f16<<<M_qkv / 16, 256, 0, stream>>>(Vin16, Wv16, bv, V16, nullptr, nullptr, nullptr, nullptr, M_qkv);
    k_proj_f16<<<M_pos / 16, 256, 0, stream>>>(Pin16, Wp16, nullptr, P16, nullptr, nullptr, nullptr, nullptr, M_pos);

    // V transpose for contiguous K in attn@V
    {
        const size_t n = (size_t)BATCH * NUM_HEADS * D_K * T2;
        k_vt<<<(unsigned)((n + 255) / 256), 256, 0, stream>>>(V16, VT16);
    }

    // Content scores (all batches/heads)
    k_scores_ac<<<4096, 256, 0, stream>>>(QU16, K16, SC);

    // Positional scores + shift + softmax, batch by batch (reuses BD scratch)
    for (int b = 0; b < BATCH; ++b) {
        k_scores_bd<<<512, 256, 0, stream>>>(QV16 + (size_t)b * T1 * EMBED,
                                             P16 + (size_t)b * N_POS * EMBED, BD);
        k_softmax<<<NUM_HEADS * T1, 256, 0, stream>>>(
            SC + (size_t)b * NUM_HEADS * T1 * T2, BD,
            mask + (size_t)b * T2, chunk,
            ATT + (size_t)b * NUM_HEADS * T1 * T2);
    }

    // Context and output projection
    k_ctx<<<512, 256, 0, stream>>>(ATT, VT16, CTX16);
    k_proj_f16<<<M_qkv / 16, 256, 0, stream>>>(CTX16, Wo16, bo, nullptr, nullptr, nullptr, nullptr, out, M_qkv);
}
